// Net_257698038039
// MI455X (gfx1250) — matrix operand order repaired and checked by reading, not compiled, not run on hardware
//
#include <hip/hip_runtime.h>
#include <hip/hip_bf16.h>
#include <stdint.h>

#define HEADS 8

typedef __attribute__((ext_vector_type(16))) _Float16 v16h;
typedef __attribute__((ext_vector_type(8)))  float    v8f;

// ---- order-preserving float<->uint encoding for atomicMax-based segment max
__device__ __forceinline__ unsigned enc_f32(float f) {
  unsigned u = __float_as_uint(f);
  return (u & 0x80000000u) ? ~u : (u | 0x80000000u);
}
__device__ __forceinline__ float dec_f32(unsigned u) {
  return __uint_as_float((u & 0x80000000u) ? (u & 0x7FFFFFFFu) : ~u);
}

// =====================================================================
// Pack W[K, NCOLS] (f32, row-major) into WMMA B-fragment-major f16:
// Bp[((t*kSteps + kb)*32 + lane)*16 + idx] = (f16) W[k, n]
//   k = kb*32 + (lane>>4)*16 + idx,  n = t*16 + (lane&15)
// so each lane's 32x16 B fragment is one contiguous 32-byte v16h.
// =====================================================================
__global__ void pack_b_kernel(const float* __restrict__ W, _Float16* __restrict__ Bp,
                              int K, int NCOLS) {
  int o = blockIdx.x * blockDim.x + threadIdx.x;
  if (o >= K * NCOLS) return;
  int idx    = o & 15;
  int lane   = (o >> 4) & 31;
  int r      = o >> 9;
  int kSteps = K >> 5;
  int kb     = r % kSteps;
  int t      = r / kSteps;
  int k = (kb << 5) + ((lane >> 4) << 4) + idx;
  int n = (t << 4) + (lane & 15);
  Bp[o] = (_Float16)W[(size_t)k * NCOLS + n];
}

// =====================================================================
// WMMA GEMM: C[M, NT*16] = A[M,K] * B[K, NT*16]; A fp32 (converted on the
// fly), B pre-swizzled f16 fragments, f32 accumulation via
// v_wmma_f32_16x16x32_f16. One wave per 16-row tile; EXEC all-ones
// (wave-uniform guard). M % 16 == 0, K % 32 == 0.
// =====================================================================
template<int NT>
__global__ __launch_bounds__(256)
void gemm_wmma_kernel(const float* __restrict__ A, const _Float16* __restrict__ Bp,
                      float* __restrict__ C, int M, int K) {
  const int NCOLS = NT * 16;
  int wave = blockIdx.x * (blockDim.x >> 5) + (threadIdx.x >> 5);
  int numTiles = M >> 4;
  if (wave >= numTiles) return;            // wave-uniform
  int lane = threadIdx.x & 31;
  int m0   = wave << 4;
  int lrow = lane & 15;
  int lhi  = lane >> 4;                    // 0 or 1 (lane half)

  v8f zero = {};
  v8f acc[NT];
#pragma unroll
  for (int t = 0; t < NT; ++t) acc[t] = zero;

  const float* arow = A + (size_t)(m0 + lrow) * K;
  const v16h* bptr  = (const v16h*)Bp;
  const int kSteps  = K >> 5;

  for (int ks = 0; ks < kSteps; ++ks) {
    int k0 = ks << 5;
    // ---- A fragment: halves 0-7 <- A[k0+16*lhi .. +7], 8-15 <- A[k0+16*lhi+8 .. +7]
    const float4* ap0 = (const float4*)(arow + k0 + (lhi << 4));
    const float4* ap1 = (const float4*)(arow + k0 + (lhi << 4) + 8);
    float4 a0 = ap0[0], a1 = ap0[1];
    float4 a2 = ap1[0], a3 = ap1[1];
    float av[16] = {a0.x, a0.y, a0.z, a0.w, a1.x, a1.y, a1.z, a1.w,
                    a2.x, a2.y, a2.z, a2.w, a3.x, a3.y, a3.z, a3.w};
    v16h afrag;
#pragma unroll
    for (int i = 0; i < 16; ++i) afrag[i] = (_Float16)av[i];

#pragma unroll
    for (int t = 0; t < NT; ++t) {
      v16h bfrag = bptr[(size_t)(t * kSteps + ks) * 32 + lane]; // 32B coalesced
      acc[t] = __builtin_amdgcn_wmma_f32_16x16x32_f16(
          false, afrag, false, bfrag, (short)0, acc[t], false, false);
    }
  }

  // ---- store C per 16x16 f32 D layout: VGPR i -> row i + 8*lanehalf
#pragma unroll
  for (int t = 0; t < NT; ++t) {
    float* crow = C + (size_t)m0 * NCOLS + t * 16 + lrow;
#pragma unroll
    for (int i = 0; i < 8; ++i) {
      int m = i + (lhi << 3);
      crow[(size_t)m * NCOLS] = acc[t][i];
    }
  }
}

// =====================================================================
// zero a word region (covers amax(uint)/den(f32)/agg(f32) contiguously)
// =====================================================================
__global__ void zero_kernel(unsigned* __restrict__ p, long long n) {
  long long i = (long long)blockIdx.x * blockDim.x + threadIdx.x;
  if (i < n) p[i] = 0u;
}

// =====================================================================
// Pass A: per (edge, head) — MX logits, alpha, leaky-relu, segment max.
// e >= E are the N self-loops.
// =====================================================================
template<int C>
__global__ void edge_alpha_kernel(const float* __restrict__ h,
    const int* __restrict__ src, const int* __restrict__ dst,
    const float* __restrict__ att_l, const float* __restrict__ att_r,
    float* __restrict__ alpha, unsigned* __restrict__ amax,
    int E, int Etot) {
  long long t = (long long)blockIdx.x * blockDim.x + threadIdx.x;
  if (t >= (long long)Etot * HEADS) return;
  int e  = (int)(t >> 3);
  int hd = (int)(t & 7);
  int s, d;
  if (e < E) { s = src[e]; d = dst[e]; } else { s = e - E; d = s; }
  const float* xj = h + ((size_t)s * HEADS + hd) * C;   // L2-resident gather
  const float* xi = h + ((size_t)d * HEADS + hd) * C;
  const float* al = att_l + hd * C;
  const float* ar = att_r + hd * C;
  float logit = 0.f, a = 0.f;
#pragma unroll
  for (int c = 0; c < C; ++c) {
    float vj = xj[c], vi = xi[c];
    logit = fmaf(vi, vj, logit);
    a = fmaf(vj, al[c], a);
    a = fmaf(vi, ar[c], a);
  }
  a = a * (1.f / (1.f + __expf(-logit)));               // * sigmoid(logits)
  a = (a > 0.f) ? a : 0.2f * a;                         // leaky_relu
  alpha[t] = a;
  atomicMax(&amax[(size_t)d * HEADS + hd], enc_f32(a)); // segment max
}

// =====================================================================
// Pass B: ex = exp(alpha - amax[dst]); den += ex; agg[dst] += ex * x_j.
// (division by den deferred to finalize — den constant per (dst,head))
// =====================================================================
template<int C>
__global__ void edge_scatter_kernel(const float* __restrict__ h,
    const int* __restrict__ src, const int* __restrict__ dst,
    const float* __restrict__ alpha, const unsigned* __restrict__ amax,
    float* __restrict__ den, float* __restrict__ agg, int E, int Etot) {
  long long t = (long long)blockIdx.x * blockDim.x + threadIdx.x;
  if (t >= (long long)Etot * HEADS) return;
  int e  = (int)(t >> 3);
  int hd = (int)(t & 7);
  int s, d;
  if (e < E) { s = src[e]; d = dst[e]; } else { s = e - E; d = s; }
  float mx = dec_f32(amax[(size_t)d * HEADS + hd]);
  float ex = __expf(alpha[t] - mx);
  atomicAdd(&den[(size_t)d * HEADS + hd], ex);
  const float* xj = h + ((size_t)s * HEADS + hd) * C;
  float* ag = agg + ((size_t)d * HEADS + hd) * C;
#pragma unroll
  for (int c = 0; c < C; ++c) atomicAdd(&ag[c], ex * xj[c]);
}

// =====================================================================
// finalize layer 1: x1 = elu(agg/den + bias1), flat over N*64
// =====================================================================
__global__ void finalize1_kernel(const float* __restrict__ agg,
    const float* __restrict__ den, const float* __restrict__ bias,
    float* __restrict__ x1, long long total) {
  long long t = (long long)blockIdx.x * blockDim.x + threadIdx.x;
  if (t >= total) return;
  float v = agg[t] / den[t >> 3] + bias[t & 63];
  x1[t] = (v > 0.f) ? v : (__expf(v) - 1.f);            // elu
}

// =====================================================================
// finalize layer 2: mean over heads of agg/den, + bias2, log_softmax.
// Also writes att_loss = 0 at out[N*16].
// =====================================================================
__global__ void finalize2_kernel(const float* __restrict__ agg,
    const float* __restrict__ den, const float* __restrict__ bias,
    float* __restrict__ out, int N) {
  int n = blockIdx.x * blockDim.x + threadIdx.x;
  if (n >= N) return;
  float inv[HEADS];
#pragma unroll
  for (int hd = 0; hd < HEADS; ++hd) inv[hd] = 1.f / den[(size_t)n * HEADS + hd];
  float vals[16];
  float mx = -3.0e38f;
#pragma unroll
  for (int c = 0; c < 16; ++c) {
    float s = 0.f;
#pragma unroll
    for (int hd = 0; hd < HEADS; ++hd)
      s += agg[((size_t)n * HEADS + hd) * 16 + c] * inv[hd];
    float v = s * 0.125f + bias[c];
    vals[c] = v;
    mx = fmaxf(mx, v);
  }
  float sum = 0.f;
#pragma unroll
  for (int c = 0; c < 16; ++c) sum += __expf(vals[c] - mx);
  float lse = mx + __logf(sum);
#pragma unroll
  for (int c = 0; c < 16; ++c) out[(size_t)n * 16 + c] = vals[c] - lse;
  if (n == 0) out[(size_t)N * 16] = 0.f;                // att_loss
}

// =====================================================================
extern "C" void kernel_launch(void* const* d_in, const int* in_sizes, int n_in,
                              void* d_out, int out_size, void* d_ws, size_t ws_size,
                              hipStream_t stream) {
  const float* x   = (const float*)d_in[0];
  const int*   ei  = (const int*)d_in[1];
  const float* W1  = (const float*)d_in[2];
  const float* al1 = (const float*)d_in[3];
  const float* ar1 = (const float*)d_in[4];
  const float* b1  = (const float*)d_in[5];
  const float* W2  = (const float*)d_in[6];
  const float* al2 = (const float*)d_in[7];
  const float* ar2 = (const float*)d_in[8];
  const float* b2  = (const float*)d_in[9];
  float* out = (float*)d_out;

  const int N    = in_sizes[0] / 512;
  const int E    = in_sizes[1] / 2;
  const int Etot = E + N;
  const int* src = ei;
  const int* dst = ei + E;

  // ---- carve workspace (all sizes 256B multiples; amax/den/agg contiguous)
  char* ws = (char*)d_ws;
  auto carve = [&](size_t bytes) {
    char* p = ws; ws += (bytes + 255) & ~(size_t)255; return p;
  };
  float*     h1    = (float*)carve((size_t)N * 64 * 4);
  float*     x1    = (float*)carve((size_t)N * 64 * 4);
  float*     h2    = (float*)carve((size_t)N * 128 * 4);
  float*     alpha = (float*)carve((size_t)Etot * HEADS * 4);
  unsigned*  amax  = (unsigned*)carve((size_t)N * HEADS * 4);
  float*     den   = (float*)carve((size_t)N * HEADS * 4);
  float*     agg   = (float*)carve((size_t)N * 128 * 4);
  _Float16*  W1p   = (_Float16*)carve((size_t)512 * 64 * 2);
  _Float16*  W2p   = (_Float16*)carve((size_t)64 * 128 * 2);

  const int TPB = 256;
  long long eth = (long long)Etot * HEADS;
  int eblocks = (int)((eth + TPB - 1) / TPB);
  int gblocks = (N / 16 + 7) / 8;                       // 8 waves per block

  // ---- pre-swizzle weights into f16 WMMA fragments (tiny)
  pack_b_kernel<<<(512 * 64 + TPB - 1) / TPB, TPB, 0, stream>>>(W1, W1p, 512, 64);
  pack_b_kernel<<<(64 * 128 + TPB - 1) / TPB, TPB, 0, stream>>>(W2, W2p, 64, 128);

  // ---------------- layer 1 ----------------
  gemm_wmma_kernel<4><<<gblocks, TPB, 0, stream>>>(x, W1p, h1, N, 512);

  long long z1 = (long long)N * HEADS * 2 + (long long)N * 64;   // amax+den+agg1
  zero_kernel<<<(int)((z1 + TPB - 1) / TPB), TPB, 0, stream>>>(amax, z1);

  edge_alpha_kernel<8><<<eblocks, TPB, 0, stream>>>(h1, src, dst, al1, ar1,
                                                    alpha, amax, E, Etot);
  edge_scatter_kernel<8><<<eblocks, TPB, 0, stream>>>(h1, src, dst, alpha, amax,
                                                      den, agg, E, Etot);
  long long nf1 = (long long)N * 64;
  finalize1_kernel<<<(int)((nf1 + TPB - 1) / TPB), TPB, 0, stream>>>(agg, den, b1,
                                                                     x1, nf1);

  // ---------------- layer 2 ----------------
  gemm_wmma_kernel<8><<<gblocks, TPB, 0, stream>>>(x1, W2p, h2, N, 64);

  long long z2 = (long long)N * HEADS * 2 + (long long)N * 128;  // amax+den+agg2
  zero_kernel<<<(int)((z2 + TPB - 1) / TPB), TPB, 0, stream>>>(amax, z2);

  edge_alpha_kernel<16><<<eblocks, TPB, 0, stream>>>(h2, src, dst, al2, ar2,
                                                     alpha, amax, E, Etot);
  edge_scatter_kernel<16><<<eblocks, TPB, 0, stream>>>(h2, src, dst, alpha, amax,
                                                       den, agg, E, Etot);
  finalize2_kernel<<<(N + TPB - 1) / TPB, TPB, 0, stream>>>(agg, den, b2, out, N);
}
